// DistiledMultiheadAttention_38500086842162
// MI455X (gfx1250) — matrix operand order repaired and checked by reading, not compiled, not run on hardware
//
#include <hip/hip_runtime.h>
#include <hip/hip_bf16.h>
#include <math.h>

// ---------------------------------------------------------------------------
// DistiledMultiheadAttention for MI455X (gfx1250, wave32, WMMA bf16)
//
//  * wk/wv projections folded through the q side (kills the O(N*48*E*E) GEMMs)
//  * all GEMMs on v_wmma_f32_16x16x32_bf16 (fp32 accumulate)
//  * weights pre-packed in WMMA fragment order; B tile staging is a pure
//    contiguous copy on global_load_async_to_lds_b128, double-buffered and
//    pipelined behind the WMMAs with s_wait_asynccnt
//  * fragment-major LDS: every operand fetch is a ds_load_b128; A staging is
//    2x global_load_b128 + 1x ds_store_b128 per thread per K-step
// ---------------------------------------------------------------------------

#define WB    4
#define EE    512
#define FF    1024
#define HH    8
#define HD    64
#define WIN   16
#define MAXL  768
#define NMAX  3072
#define UHID  160

typedef __bf16 bf16_t;
typedef __bf16 v16bf __attribute__((ext_vector_type(16)));
typedef __bf16 v8bf  __attribute__((ext_vector_type(8)));
typedef float  v8f   __attribute__((ext_vector_type(8)));

// ---------------- CDNA5 async copy helpers ----------------

__device__ __forceinline__ uint32_t lds_off(const void* p) {
  // generic LDS pointer: addr[31:0] is the LDS offset (aperture lives in [63:32])
  return (uint32_t)(uintptr_t)p;
}

__device__ __forceinline__ void async_copy_b128(uint32_t lds, const void* g) {
  unsigned long long ga = (unsigned long long)g;
  asm volatile("global_load_async_to_lds_b128 %0, %1, off"
               :: "v"(lds), "v"(ga) : "memory");
}

__device__ __forceinline__ void wait_async0() {
  asm volatile("s_wait_asynccnt 0" ::: "memory");
}
__device__ __forceinline__ void wait_async2() {
  asm volatile("s_wait_asynccnt 2" ::: "memory");
}

// ---------------- small device helpers ----------------

__device__ __forceinline__ void findb(const int* __restrict__ starts, int row, int& b, int& qp) {
  b = 0;
#pragma unroll
  for (int i = 1; i < WB; ++i) if (row >= starts[i]) b = i;
  qp = row - starts[b];
}

__device__ __forceinline__ float geluf(float x) {
  return 0.5f * x * (1.0f + erff(x * 0.70710678118654752f));
}

// ---------------- meta: starts from lengths ----------------
// meta ints: [0..4] starts0, [5] n_total, [6..10] starts1, [11] n1,
//            [12..16] starts2, [17] n2
__global__ void k_meta(const int* __restrict__ lengths, int* __restrict__ meta) {
  if (threadIdx.x == 0 && blockIdx.x == 0) {
    int acc = 0;
    for (int b = 0; b < WB; ++b) { meta[b] = acc; acc += lengths[b]; }
    meta[WB] = acc;
    meta[5]  = acc;
  }
}

// ---------------- weight pack: fp32 [K x Nc] -> bf16 fragment order ----------
__global__ void k_pack(const float* __restrict__ src, int lds, int rowOff, int colOff,
                       int K, int Nc, int transpose, bf16_t* __restrict__ dst, int total) {
  int idx = blockIdx.x * 256 + threadIdx.x;
  if (idx >= total) return;
  int KT = K >> 6;
  int blk = idx >> 12, inner = idx & 4095;
  int bn = blk / KT, kt = blk % KT;
  int kc = inner >> 11, rem = inner & 2047;
  int sub = rem >> 9, rem2 = rem & 511;
  int col = rem2 >> 5, kk = rem2 & 31;
  int srcK = kt * 64 + kc * 32 + kk;
  int srcN = bn * 64 + sub * 16 + col;
  float v = 0.f;
  if (srcN < Nc)
    v = transpose ? src[(size_t)(colOff + srcN) * lds + rowOff + srcK]
                  : src[(size_t)(rowOff + srcK) * lds + colOff + srcN];
  dst[idx] = (bf16_t)v;
}

// ---------------- flexible WMMA GEMM ----------------

struct GemmArgs {
  const int*    mcnt;
  int           K, Nc;
  int           amode;      // 0 dense, 1 x-flat gather, 2 window gather
  const float*  A;
  int           lda, acol;
  const float*  wbuf;
  const int*    sel;
  const int*    astarts;
  const bf16_t* Wp;         // fragment-packed weights
  const float*  bias;
  float*        C;
  int           ldc, ccol;
  int           crowmap;
  const int*    cstarts;
};

// pointer to 8 contiguous fp32 A elements (all modes keep groups of 8 contiguous)
__device__ __forceinline__ const float* gemm_ptrA8(const GemmArgs& g, int row, int k) {
  if (g.amode == 0) {
    return &g.A[(size_t)row * g.lda + g.acol + k];
  } else if (g.amode == 1) {
    int b, qp; findb(g.astarts, row, b, qp);
    return &g.A[((size_t)b * MAXL + qp) * EE + k];
  } else {
    int r = g.sel ? g.sel[row] : row;
    int j = k >> 10;
    int f = k & (FF - 1);
    int b, qp; findb(g.astarts, r, b, qp);
    int src = qp - j;
    return (src >= 0) ? &g.A[(size_t)(g.astarts[b] + src) * FF + f]
                      : &g.wbuf[(size_t)(WIN - 1 + src) * FF + f];
  }
}

#define BM 32
#define BN 64
#define BK 64

__global__ __launch_bounds__(256) void k_gemm(GemmArgs g) {
  const int M = *g.mcnt;
  const int bm = blockIdx.y, bn = blockIdx.x;
  if (bm * BM >= M) return;

  // fragment-major tiles: each lane's operand = 16 contiguous bf16 (32 B)
  __shared__ __align__(32) bf16_t As2[2 * 2 * 32 * 16];  // [kc][wmTile][lane][16]
  __shared__ __align__(32) bf16_t Bs2[2][4096];          // double-buffered packed B

  const int tid = threadIdx.x;
  const int wave = tid >> 5, lane = tid & 31;
  const int wm = wave >> 2, wn = wave & 3;   // 2 x 4 wave tiles of 16x16

  const bf16_t* wblk = g.Wp + (size_t)bn * (g.K >> 6) * 4096;
  const uint32_t bsBase = lds_off(Bs2) + tid * 32;

  // A staging task (one per thread): row r, k-group of 8 -> one 16 B LDS store
  const int ar   = tid >> 3;
  const int akb8 = (tid & 7) * 8;
  const int akc = akb8 >> 5, akloc = akb8 & 31;
  const int apos0 = akloc & 15;
  const int aln = (ar & 15) + 16 * ((akloc >> 4) & 1);
  bf16_t* aDst = &As2[((((akc * 2 + (ar >> 4)) * 32) + aln) * 16) + apos0];
  const int arow = bm * BM + ar;

  v8f acc = {0.f, 0.f, 0.f, 0.f, 0.f, 0.f, 0.f, 0.f};

  // prologue: kick off first B tile
  {
    const char* gb = (const char*)wblk + tid * 32;
    async_copy_b128(bsBase, gb);
    async_copy_b128(bsBase + 16, gb + 16);
  }

  for (int k0 = 0, t = 0; k0 < g.K; k0 += BK, ++t) {
    const int par = t & 1;
    const bool hasNext = (k0 + BK < g.K);

    // issue next B tile into the other buffer (overlaps with A stage + WMMA)
    if (hasNext) {
      const char* gb = (const char*)wblk + 4096 * 2 /*bytes per blk*/ + tid * 32;
      uint32_t nb = bsBase + ((par ^ 1) ? 8192u : 0u) - (par ? 8192u : 0u);
      // nb = base of buffer (par^1) + tid*32
      nb = lds_off(Bs2) + (uint32_t)(par ^ 1) * 8192u + tid * 32;
      async_copy_b128(nb, gb);
      async_copy_b128(nb + 16, gb + 16);
      __builtin_prefetch((const char*)wblk + 16384 + tid * 32, 0, 1);
    }

    // --- A tile: 2x global b128 + cvt + 1x ds_store_b128 per thread ---
    {
      v8bf vals;
      if (arow < M) {
        const float* p = gemm_ptrA8(g, arow, k0 + akb8);
        const float4 lo = ((const float4*)p)[0];
        const float4 hi = ((const float4*)p)[1];
        vals[0] = (bf16_t)lo.x; vals[1] = (bf16_t)lo.y;
        vals[2] = (bf16_t)lo.z; vals[3] = (bf16_t)lo.w;
        vals[4] = (bf16_t)hi.x; vals[5] = (bf16_t)hi.y;
        vals[6] = (bf16_t)hi.z; vals[7] = (bf16_t)hi.w;
      } else {
#pragma unroll
        for (int i = 0; i < 8; ++i) vals[i] = (bf16_t)0.f;
      }
      *(v8bf*)aDst = vals;
    }

    // current tile's 2 async ops retired; next tile's may remain in flight
    if (hasNext) wait_async2(); else wait_async0();
    __syncthreads();

#pragma unroll
    for (int kc = 0; kc < 2; ++kc) {
      v16bf af = *(const v16bf*)&As2[(((kc * 2 + wm) * 32) + lane) * 16];
      const int kb = (lane < 16) ? 0 : 16;
      v16bf bfr = *(const v16bf*)&Bs2[par][((kc * 4 + wn) * 16 + (lane & 15)) * 32 + kb];
      acc = __builtin_amdgcn_wmma_f32_16x16x32_bf16(false, af, false, bfr,
                                                    (short)0, acc, false, false);
    }
    __syncthreads();
    wblk += 4096;
  }

  // epilogue: f32 C 16x16 layout (VGPR r: lanes0-15 M=r, lanes16-31 M=8+r)
  const int ncol = bn * BN + wn * 16 + (lane & 15);
  if (ncol < g.Nc) {
    const float bia = g.bias ? g.bias[ncol] : 0.f;
    const int mbase = bm * BM + wm * 16 + ((lane < 16) ? 0 : 8);
#pragma unroll
    for (int r = 0; r < 8; ++r) {
      int row = mbase + r;
      if (row < M) {
        int crow = row;
        if (g.crowmap) {
          int b, qp; findb(g.cstarts, row, b, qp);
          crow = b * MAXL + qp;
        }
        g.C[(size_t)crow * g.ldc + g.ccol + ncol] = acc[r] + bia;
      }
    }
  }
}

// ---------------- utility MLP head: eu = elu(gelu(h)@w2 + b2) + 1 ----------------

__global__ __launch_bounds__(256) void k_uhead(const float* __restrict__ h,
                                               const float* __restrict__ w2,
                                               const float* __restrict__ b2,
                                               const int* __restrict__ mcnt,
                                               float* __restrict__ eu) {
  int row = blockIdx.x * 8 + (threadIdx.x >> 5);
  int lane = threadIdx.x & 31;
  if (row >= *mcnt) return;
  float s = 0.f;
  for (int k = lane; k < UHID; k += 32) s += geluf(h[(size_t)row * UHID + k]) * w2[k];
#pragma unroll
  for (int o = 16; o; o >>= 1) s += __shfl_xor(s, o, 32);
  if (lane == 0) {
    float x = s + b2[0];
    eu[row] = ((x > 0.f) ? x : (expf(x) - 1.f)) + 1.f;
  }
}

// ---------------- distill head: r = softmax(gelu(h)@w2 + b2) over 16 ----------------

__global__ __launch_bounds__(256) void k_dhead(const float* __restrict__ h,
                                               const float* __restrict__ w2,
                                               const float* __restrict__ b2,
                                               const int* __restrict__ mcnt,
                                               float* __restrict__ r) {
  int row = blockIdx.x * 8 + (threadIdx.x >> 5);
  int lane = threadIdx.x & 31;
  if (row >= *mcnt) return;
  float lg = -1e30f;
  if (lane < WIN) {
    float s = b2[lane];
    for (int k = 0; k < UHID; ++k) s += geluf(h[(size_t)row * UHID + k]) * w2[k * WIN + lane];
    lg = s;
  }
  float mx = lg;
#pragma unroll
  for (int o = 8; o; o >>= 1) mx = fmaxf(mx, __shfl_xor(mx, o, 16));
  float e = (lane < WIN) ? expf(lg - mx) : 0.f;
  float sum = e;
#pragma unroll
  for (int o = 8; o; o >>= 1) sum += __shfl_xor(sum, o, 16);
  if (lane < WIN) r[(size_t)row * WIN + lane] = e / sum;
}

// ---------------- serial selection scan (literal reference semantics) ----------------

__global__ void k_select(const float* __restrict__ eu, const int* __restrict__ stin,
                         int* __restrict__ selidx, int* __restrict__ tm,
                         int* __restrict__ stout) {
  if (threadIdx.x || blockIdx.x) return;
  float S[WB];
  for (int b = 0; b < WB; ++b) {
    float s = 0.f;
    for (int n = stin[b]; n < stin[b + 1]; ++n) s += eu[n];
    S[b] = s;
  }
  float off[WB];
  off[0] = 0.f; off[1] = S[0]; off[2] = S[0] + S[2]; off[3] = S[0] + S[2] + S[3];
  float cum = 0.f;
  int cnt = 0, csel0 = 0;
  int lens[WB];
  for (int b = 0; b < WB; ++b) {
    int before = cnt;
    for (int n = stin[b]; n < stin[b + 1]; ++n) {
      cum += eu[n];
      float running = cum - off[b];
      float idxs = (float)(n + 1 - stin[b]);
      bool selg = (eu[n] > running / idxs) || (idxs <= (float)WIN);
      if (selg) { selidx[cnt] = n; cnt++; }
      if (n == 0) csel0 = cnt;
      tm[n] = cnt - csel0;
    }
    lens[b] = cnt - before;
  }
  int acc = 0;
  for (int b = 0; b < WB; ++b) { stout[b] = acc; acc += lens[b]; }
  stout[WB] = acc;
  stout[5]  = acc;
}

// ---------------- dat = sum_j window(sel[row], j) * r[row][j] ----------------

__global__ __launch_bounds__(256) void k_dat(const float* __restrict__ data,
                                             const float* __restrict__ buf,
                                             const int* __restrict__ starts,
                                             const int* __restrict__ sel,
                                             const float* __restrict__ r,
                                             const int* __restrict__ mcnt,
                                             float* __restrict__ out) {
  int row = blockIdx.x;
  int f = blockIdx.y * 256 + threadIdx.x;
  if (row >= *mcnt) return;
  __shared__ float rw[WIN];
  if (threadIdx.x < WIN) rw[threadIdx.x] = r[(size_t)row * WIN + threadIdx.x];
  __syncthreads();
  int base = sel[row];
  int b, qp; findb(starts, base, b, qp);
  float acc = 0.f;
#pragma unroll
  for (int j = 0; j < WIN; ++j) {
    int src = qp - j;
    float v = (src >= 0) ? data[(size_t)(starts[b] + src) * FF + f]
                         : buf[(size_t)(WIN - 1 + src) * FF + f];
    acc += v * rw[j];
  }
  out[(size_t)row * FF + f] = acc;
}

// ---------------- fused per-token scores / softmax / vmix ----------------

struct ScoreArgs {
  const int*   meta;
  const float* T;     // [N][H*E]
  const float* qh;    // [N][E]
  const float* bk;    // [E]
  const float* kv0;  const float* dat1;  const float* dat2;
  const float* buf0; const float* buf1;  const float* buf2;
  const float* eu0;  const float* eu1;
  const int*   sel0; const int*   sel1;
  const int*   tm0;  const int*   tm1;
  float*       vmix;  // [N][H*E]
};

__device__ __forceinline__ const float* resolve_row(const ScoreArgs& a, int n, int s,
                                                    int i1, int i2, float* scale) {
  int j, r;
  const float *data, *buf, *eu = nullptr;
  const int *st, *sel = nullptr;
  if (s < WIN)          { j = s;           r = n;  data = a.kv0;  buf = a.buf0; st = a.meta;      }
  else if (s < 2 * WIN) { j = s - WIN;     r = i1; data = a.dat1; buf = a.buf1; st = a.meta + 6;  eu = a.eu0; sel = a.sel0; }
  else                  { j = s - 2 * WIN; r = i2; data = a.dat2; buf = a.buf2; st = a.meta + 12; eu = a.eu1; sel = a.sel1; }
  int b, qp; findb(st, r, b, qp);
  int src = qp - j;
  *scale = eu ? ((src >= 0) ? eu[sel[st[b] + src]] : 0.f) : 1.f;
  return (src >= 0) ? data + (size_t)(st[b] + src) * FF
                    : buf + (size_t)(WIN - 1 + src) * FF;
}

__global__ __launch_bounds__(256) void k_scores(ScoreArgs a) {
  __shared__ __align__(16) float Tl[HH * EE];   // 16 KB
  __shared__ float sb[HH];
  __shared__ float sc[HH][3 * WIN];
  const int n = blockIdx.x;
  if (n >= a.meta[5]) return;
  const int tid = threadIdx.x, wave = tid >> 5, lane = tid & 31;

  // async copy of the folded-key row T[n] (16 KB contiguous) into LDS
  {
    const char* gp = (const char*)(a.T + (size_t)n * (HH * EE)) + tid * 16;
    uint32_t lb = lds_off(Tl) + tid * 16;
#pragma unroll
    for (int i = 0; i < 4; ++i)
      async_copy_b128(lb + i * 4096, gp + i * 4096);
  }

  { // sbias[h] = qh_h . bk_h
    float s = 0.f;
    const float* q = a.qh + (size_t)n * EE + wave * HD;
    for (int c = lane; c < HD; c += 32) s += q[c] * a.bk[wave * HD + c];
#pragma unroll
    for (int o = 16; o; o >>= 1) s += __shfl_xor(s, o, 32);
    if (lane == 0) sb[wave] = s;
  }
  wait_async0();
  __syncthreads();

  const int i1 = a.tm0[n];
  const int i2 = a.tm1[i1];

  for (int s = wave; s < 3 * WIN; s += HH) {
    float scale;
    const float* rp = resolve_row(a, n, s, i1, i2, &scale);
#pragma unroll
    for (int h = 0; h < HH; ++h) {
      float acc = 0.f;
      for (int k = lane; k < EE; k += 32) acc += rp[k] * Tl[h * EE + k];
#pragma unroll
      for (int o = 16; o; o >>= 1) acc += __shfl_xor(acc, o, 32);
      if (lane == 0) sc[h][s] = (acc * scale + sb[h]) * 0.125f;
    }
  }
  __syncthreads();

  if (lane == 0) {
    float mx = -1e30f;
    for (int s = 0; s < 3 * WIN; ++s) mx = fmaxf(mx, sc[wave][s]);
    float sum = 0.f;
    for (int s = 0; s < 3 * WIN; ++s) { float e = expf(sc[wave][s] - mx); sc[wave][s] = e; sum += e; }
    float inv = 1.f / sum;
    for (int s = 0; s < 3 * WIN; ++s) sc[wave][s] *= inv;
  }
  __syncthreads();

  float acc[16];
#pragma unroll
  for (int i = 0; i < 16; ++i) acc[i] = 0.f;
  for (int s = 0; s < 3 * WIN; ++s) {
    float scale;
    const float* rp = resolve_row(a, n, s, i1, i2, &scale);
    float w = sc[wave][s] * scale;
    const float* vp = rp + EE + lane * 16;
#pragma unroll
    for (int i = 0; i < 16; ++i) acc[i] += w * vp[i];
  }
  float* o = a.vmix + (size_t)n * (HH * EE) + wave * EE + lane * 16;
#pragma unroll
  for (int i = 0; i < 16; ++i) o[i] = acc[i];
}

// ---------------------------------------------------------------------------
// host side
// ---------------------------------------------------------------------------

static void launch_gemm(hipStream_t st, const int* mcnt, int Mmax, int K, int Nc,
                        int amode, const float* A, int lda, int acol,
                        const float* wbuf, const int* sel, const int* astarts,
                        const bf16_t* Wp,
                        const float* bias, float* C, int ldc, int ccol,
                        int crowmap, const int* cstarts) {
  GemmArgs g;
  g.mcnt = mcnt; g.K = K; g.Nc = Nc;
  g.amode = amode; g.A = A; g.lda = lda; g.acol = acol;
  g.wbuf = wbuf; g.sel = sel; g.astarts = astarts;
  g.Wp = Wp;
  g.bias = bias; g.C = C; g.ldc = ldc; g.ccol = ccol;
  g.crowmap = crowmap; g.cstarts = cstarts;
  dim3 grid((Nc + BN - 1) / BN, (Mmax + BM - 1) / BM);
  k_gemm<<<grid, dim3(256), 0, st>>>(g);
}

extern "C" void kernel_launch(void* const* d_in, const int* in_sizes, int n_in,
                              void* d_out, int out_size, void* d_ws, size_t ws_size,
                              hipStream_t stream) {
  (void)in_sizes; (void)n_in; (void)ws_size;
  const float* x      = (const float*)d_in[0];
  const int*   lens   = (const int*)  d_in[1];
  const float* kv_w   = (const float*)d_in[2];
  const float* kv_b   = (const float*)d_in[3];
  const float* q_w    = (const float*)d_in[4];
  const float* q_b    = (const float*)d_in[5];
  const float* proj_w = (const float*)d_in[6];
  const float* proj_b = (const float*)d_in[7];
  const float* buf0   = (const float*)d_in[8];
  const float* buf1   = (const float*)d_in[9];
  const float* buf2   = (const float*)d_in[10];
  const float* d0_w1  = (const float*)d_in[11];
  const float* d0_b1  = (const float*)d_in[12];
  const float* d0_w2  = (const float*)d_in[13];
  const float* d0_b2  = (const float*)d_in[14];
  const float* u0_w1  = (const float*)d_in[15];
  const float* u0_b1  = (const float*)d_in[16];
  const float* u0_w2  = (const float*)d_in[17];
  const float* u0_b2  = (const float*)d_in[18];
  const float* d1_w1  = (const float*)d_in[19];
  const float* d1_b1  = (const float*)d_in[20];
  const float* d1_w2  = (const float*)d_in[21];
  const float* d1_b2  = (const float*)d_in[22];
  const float* u1_w1  = (const float*)d_in[23];
  const float* u1_b1  = (const float*)d_in[24];
  const float* u1_w2  = (const float*)d_in[25];
  const float* u1_b2  = (const float*)d_in[26];
  const float* wq     = (const float*)d_in[27];
  const float* bq     = (const float*)d_in[28];
  const float* wk     = (const float*)d_in[29];
  const float* bk     = (const float*)d_in[30];
  const float* wv     = (const float*)d_in[31];
  const float* bv     = (const float*)d_in[32];
  const float* out_w  = (const float*)d_in[33];
  const float* out_b  = (const float*)d_in[34];

  char* wp = (char*)d_ws;
  auto alloc = [&](size_t bytes) -> void* {
    void* p = (void*)wp;
    wp += (bytes + 255) & ~(size_t)255;
    return p;
  };
  int*   meta  = (int*)  alloc(64 * sizeof(int));
  int*   sel0  = (int*)  alloc(NMAX * sizeof(int));
  int*   tm0   = (int*)  alloc(NMAX * sizeof(int));
  int*   sel1  = (int*)  alloc(NMAX * sizeof(int));
  int*   tm1   = (int*)  alloc(NMAX * sizeof(int));
  float* kv0   = (float*)alloc((size_t)NMAX * FF * 4);
  float* dat1  = (float*)alloc((size_t)NMAX * FF * 4);
  float* dat2  = (float*)alloc((size_t)NMAX * FF * 4);
  float* eu0   = (float*)alloc(NMAX * 4);
  float* eu1   = (float*)alloc(NMAX * 4);
  float* hbuf  = (float*)alloc((size_t)NMAX * UHID * 4);
  float* r0    = (float*)alloc((size_t)NMAX * WIN * 4);
  float* r1    = (float*)alloc((size_t)NMAX * WIN * 4);
  float* qbuf  = (float*)alloc((size_t)NMAX * EE * 4);
  float* qh    = (float*)alloc((size_t)NMAX * EE * 4);
  float* ctx   = (float*)alloc((size_t)NMAX * EE * 4);
  float* y1    = (float*)alloc((size_t)NMAX * EE * 4);
  float* Tbuf  = (float*)alloc((size_t)NMAX * HH * EE * 4);
  float* vmix  = (float*)alloc((size_t)NMAX * HH * EE * 4);
  auto psize = [](int K, int Nc) { return (size_t)((Nc + 63) / 64) * (K >> 6) * 4096; };
  bf16_t* kvw_p   = (bf16_t*)alloc(psize(EE, FF) * 2);
  bf16_t* qw_p    = (bf16_t*)alloc(psize(EE, EE) * 2);
  bf16_t* wq_p    = (bf16_t*)alloc(psize(EE, EE) * 2);
  bf16_t* wkT_p   = (bf16_t*)alloc((size_t)HH * psize(HD, EE) * 2);
  bf16_t* wv_p    = (bf16_t*)alloc((size_t)HH * psize(EE, HD) * 2);
  bf16_t* outw_p  = (bf16_t*)alloc(psize(EE, EE) * 2);
  bf16_t* projw_p = (bf16_t*)alloc(psize(EE, EE) * 2);
  const int KW = WIN * FF;
  bf16_t* u0w1_p  = (bf16_t*)alloc(psize(KW, UHID) * 2);
  bf16_t* d0w1_p  = (bf16_t*)alloc(psize(KW, UHID) * 2);
  bf16_t* u1w1_p  = (bf16_t*)alloc(psize(KW, UHID) * 2);
  bf16_t* d1w1_p  = (bf16_t*)alloc(psize(KW, UHID) * 2);

  const int* ntot_p = meta + 5;
  const int* n1_p   = meta + 11;
  const int* n2_p   = meta + 17;
  const int* st0    = meta;
  const int* st1    = meta + 6;

  hipMemsetAsync(d_out, 0, (size_t)out_size * sizeof(float), stream);
  k_meta<<<1, 32, 0, stream>>>(lens, meta);

  auto pack = [&](const float* src, int lds, int rowOff, int colOff,
                  int K, int Nc, int transpose, bf16_t* dst) {
    int total = (int)psize(K, Nc);
    k_pack<<<(total + 255) / 256, 256, 0, stream>>>(src, lds, rowOff, colOff,
                                                    K, Nc, transpose, dst, total);
  };
  pack(kv_w, FF, 0, 0, EE, FF, 0, kvw_p);
  pack(q_w, EE, 0, 0, EE, EE, 0, qw_p);
  pack(wq, EE, 0, 0, EE, EE, 0, wq_p);
  pack(out_w, EE, 0, 0, EE, EE, 0, outw_p);
  pack(proj_w, EE, 0, 0, EE, EE, 0, projw_p);
  for (int h = 0; h < HH; ++h) {
    pack(wk, EE, h * HD, 0, HD, EE, 1, wkT_p + (size_t)h * psize(HD, EE));
    pack(wv, EE, 0, h * HD, EE, HD, 0, wv_p + (size_t)h * psize(EE, HD));
  }
  pack(u0_w1, UHID, 0, 0, KW, UHID, 0, u0w1_p);
  pack(d0_w1, UHID, 0, 0, KW, UHID, 0, d0w1_p);
  pack(u1_w1, UHID, 0, 0, KW, UHID, 0, u1w1_p);
  pack(d1_w1, UHID, 0, 0, KW, UHID, 0, d1w1_p);

  // level 0
  launch_gemm(stream, ntot_p, NMAX, EE, FF, 1, x, 0, 0, nullptr, nullptr, st0,
              kvw_p, kv_b, kv0, FF, 0, 0, nullptr);
  launch_gemm(stream, ntot_p, NMAX, KW, UHID, 2, kv0, 0, 0, buf0, nullptr, st0,
              u0w1_p, u0_b1, hbuf, UHID, 0, 0, nullptr);
  k_uhead<<<(NMAX + 7) / 8, 256, 0, stream>>>(hbuf, u0_w2, u0_b2, ntot_p, eu0);
  k_select<<<1, 32, 0, stream>>>(eu0, meta, sel0, tm0, meta + 6);

  launch_gemm(stream, n1_p, NMAX, KW, UHID, 2, kv0, 0, 0, buf0, sel0, st0,
              d0w1_p, d0_b1, hbuf, UHID, 0, 0, nullptr);
  k_dhead<<<(NMAX + 7) / 8, 256, 0, stream>>>(hbuf, d0_w2, d0_b2, n1_p, r0);
  k_dat<<<dim3(NMAX, FF / 256), 256, 0, stream>>>(kv0, buf0, st0, sel0, r0, n1_p, dat1);

  // level 1
  launch_gemm(stream, n1_p, NMAX, KW, UHID, 2, dat1, 0, 0, buf1, nullptr, st1,
              u1w1_p, u1_b1, hbuf, UHID, 0, 0, nullptr);
  k_uhead<<<(NMAX + 7) / 8, 256, 0, stream>>>(hbuf, u1_w2, u1_b2, n1_p, eu1);
  k_select<<<1, 32, 0, stream>>>(eu1, meta + 6, sel1, tm1, meta + 12);

  launch_gemm(stream, n2_p, NMAX, KW, UHID, 2, dat1, 0, 0, buf1, sel1, st1,
              d1w1_p, d1_b1, hbuf, UHID, 0, 0, nullptr);
  k_dhead<<<(NMAX + 7) / 8, 256, 0, stream>>>(hbuf, d1_w2, d1_b2, n2_p, r1);
  k_dat<<<dim3(NMAX, FF / 256), 256, 0, stream>>>(dat1, buf1, st1, sel1, r1, n2_p, dat2);

  // query path
  launch_gemm(stream, ntot_p, NMAX, EE, EE, 1, x, 0, 0, nullptr, nullptr, st0,
              qw_p, q_b, qbuf, EE, 0, 0, nullptr);
  launch_gemm(stream, ntot_p, NMAX, EE, EE, 0, qbuf, EE, 0, nullptr, nullptr, nullptr,
              wq_p, bq, qh, EE, 0, 0, nullptr);

  // folded key projection: T[n, h*E + k] = qh_h[n] . wk[k, h*64+c]
  for (int h = 0; h < HH; ++h)
    launch_gemm(stream, ntot_p, NMAX, HD, EE, 0, qh, EE, h * HD, nullptr, nullptr, nullptr,
                wkT_p + (size_t)h * psize(HD, EE), nullptr,
                Tbuf, HH * EE, h * EE, 0, nullptr);

  {
    ScoreArgs a;
    a.meta = meta; a.T = Tbuf; a.qh = qh; a.bk = bk;
    a.kv0 = kv0; a.dat1 = dat1; a.dat2 = dat2;
    a.buf0 = buf0; a.buf1 = buf1; a.buf2 = buf2;
    a.eu0 = eu0; a.eu1 = eu1;
    a.sel0 = sel0; a.sel1 = sel1; a.tm0 = tm0; a.tm1 = tm1;
    a.vmix = vmix;
    k_scores<<<NMAX, 256, 0, stream>>>(a);
  }

  // ctx[n, h*64+c] = vmix_h[n] @ wv[:, h*64+c] + bv
  for (int h = 0; h < HH; ++h)
    launch_gemm(stream, ntot_p, NMAX, EE, HD, 0, vmix, HH * EE, h * EE, nullptr, nullptr, nullptr,
                wv_p + (size_t)h * psize(EE, HD), bv + h * HD, ctx, EE, h * HD, 0, nullptr);

  launch_gemm(stream, ntot_p, NMAX, EE, EE, 0, ctx, EE, 0, nullptr, nullptr, nullptr,
              outw_p, out_b, y1, EE, 0, 0, nullptr);
  launch_gemm(stream, ntot_p, NMAX, EE, EE, 0, y1, EE, 0, nullptr, nullptr, nullptr,
              projw_p, proj_b, (float*)d_out, EE, 0, 1, meta);
}
